// PNA_35966056137428
// MI455X (gfx1250) — matrix operand order repaired and checked by reading, not compiled, not run on hardware
//
#include <hip/hip_runtime.h>
#include <cmath>

// ---------------------------------------------------------------------------
// Types for CDNA5 WMMA (wave32, 16x16x32 bf16 -> f32 accumulate)
// ---------------------------------------------------------------------------
typedef __bf16 bf16_t;
typedef __attribute__((ext_vector_type(16))) __bf16 v16bf;
typedef __attribute__((ext_vector_type(8)))  float  v8f;

union FragU {            // 32 bytes: two 16B global loads -> one WMMA operand
    uint4 u[2];
    v16bf v;
};

// ---------------------------------------------------------------------------
// Kernel 1: PNA aggregation.  x:[B,N,D] f32 -> h:[B,12D] bf16
//   aggr = [mean, max, min, std(unbiased)]  (each D wide)
//   h    = [aggr, aggr*(log(4D+1)/3), aggr*(3/log(4D+1))]
// One block per b, one thread per d.  Coalesced streaming over N.
// ---------------------------------------------------------------------------
__global__ __launch_bounds__(256)
void pna_aggregate(const float* __restrict__ x, bf16_t* __restrict__ h,
                   int Nn, int D)
{
    const int b = blockIdx.x;
    const int d = threadIdx.x;

    const float* px = x + (size_t)b * Nn * D + d;

    float s  = 0.0f;
    float sq = 0.0f;
    float mx = -3.402823466e38f;
    float mn =  3.402823466e38f;

#pragma unroll 4
    for (int n = 0; n < Nn; ++n) {
        float v = px[(size_t)n * D];
        s  += v;
        sq += v * v;
        mx  = fmaxf(mx, v);
        mn  = fminf(mn, v);
    }

    const float invN = 1.0f / (float)Nn;
    const float mean = s * invN;
    float var = (sq - s * mean) / (float)(Nn - 1);
    var = fmaxf(var, 0.0f);
    const float sd = sqrtf(var);

    const float log_c = logf((float)(4 * D) + 1.0f);   // log(4D + 1)
    const float ampS  = log_c / 3.0f;                  // DELTA = 3
    const float attS  = 3.0f / log_c;

    const float a4[4]    = { mean, mx, mn, sd };
    const float scale[3] = { 1.0f, ampS, attS };

    bf16_t* ph = h + (size_t)b * (12 * D) + d;
#pragma unroll
    for (int j = 0; j < 3; ++j)
#pragma unroll
        for (int g = 0; g < 4; ++g)
            ph[(size_t)(j * 4 + g) * D] = (bf16_t)(a4[g] * scale[j]);
}

// ---------------------------------------------------------------------------
// Kernel 2: tiled transpose + f32 -> bf16 convert.
//   src: [K, N] f32 (row-major)  ->  dst: [N, K] bf16 (row-major)
// block = (32, 8), one 32x32 tile per block, LDS bounce for coalescing.
// ---------------------------------------------------------------------------
__global__ __launch_bounds__(256)
void transpose_cvt_bf16(const float* __restrict__ src, bf16_t* __restrict__ dst,
                        int K, int N)
{
    __shared__ float tile[32][33];
    const int nBase = blockIdx.x * 32;
    const int kBase = blockIdx.y * 32;

    for (int i = threadIdx.y; i < 32; i += 8)
        tile[i][threadIdx.x] = src[(size_t)(kBase + i) * N + nBase + threadIdx.x];
    __syncthreads();
    for (int i = threadIdx.y; i < 32; i += 8)
        dst[(size_t)(nBase + i) * K + kBase + threadIdx.x] =
            (bf16_t)tile[threadIdx.x][i];
}

// ---------------------------------------------------------------------------
// Kernel 3: bf16 GEMM with f32 accumulate via v_wmma_f32_16x16x32_bf16.
//   C[M,N] = A[M,K] * BT[N,K]^T + bias  (optionally ReLU + bf16 store)
// Block: 256 threads = 8 waves as 2(M) x 4(N);  wave tile = 32M x 64N
//   -> block tile 64M x 256N.  Per k-step: 2 A frags, 4 B frags, 8 WMMAs.
//
// A fragment (16-bit A 16x32 layout): lane L, row = M + (L&15),
//   K chunk at +((L>>4)*16): one contiguous 32B load (two 16B loads).
// B fragment (16-bit B 32x16 layout): lane L, col = N + (L&15),
//   K chunk at +((L>>4)*16): one contiguous 32B load (two 16B loads).
// ---------------------------------------------------------------------------
template <bool RELU_BF16_OUT>
__global__ __launch_bounds__(256)
void gemm_bf16_wmma(const bf16_t* __restrict__ A,   // [M, K]
                    const bf16_t* __restrict__ BT,  // [N, K]
                    const float*  __restrict__ bias,// [N]
                    void* __restrict__ Cout,        // bf16 [M,N] or f32 [M,N]
                    int M, int N, int K)
{
    const int lane  = threadIdx.x & 31;
    const int wave  = threadIdx.x >> 5;
    const int waveM = wave & 1;   // 0..1
    const int waveN = wave >> 1;  // 0..3

    const int mBase = blockIdx.x * 64  + waveM * 32;
    const int nBase = blockIdx.y * 256 + waveN * 64;

    const int half  = lane >> 4;        // 0 | 1
    const int lid   = lane & 15;

    const int aRow0 = mBase + lid;
    const int aKsel = half << 4;        // 0 or 16
    const int bCol0 = nBase + lid;
    const int bKsel = half << 4;        // 0 or 16

    v8f acc[2][4];
#pragma unroll
    for (int mi = 0; mi < 2; ++mi)
#pragma unroll
        for (int ni = 0; ni < 4; ++ni)
            acc[mi][ni] = (v8f){0.f, 0.f, 0.f, 0.f, 0.f, 0.f, 0.f, 0.f};

    for (int k = 0; k < K; k += 32) {
        FragU a[2];
#pragma unroll
        for (int mi = 0; mi < 2; ++mi) {
            const bf16_t* pa = A + (size_t)(aRow0 + mi * 16) * K + k + aKsel;
            a[mi].u[0] = *(const uint4*)(pa);        // K: sel .. sel+7
            a[mi].u[1] = *(const uint4*)(pa + 8);    // K: sel+8 .. sel+15
        }
        FragU bfr[4];
#pragma unroll
        for (int ni = 0; ni < 4; ++ni) {
            const bf16_t* pb = BT + (size_t)(bCol0 + ni * 16) * K + k + bKsel;
            bfr[ni].u[0] = *(const uint4*)(pb);      // K: sel .. sel+7
            bfr[ni].u[1] = *(const uint4*)(pb + 8);  // K: sel+8 .. sel+15
        }
#pragma unroll
        for (int mi = 0; mi < 2; ++mi)
#pragma unroll
            for (int ni = 0; ni < 4; ++ni)
                acc[mi][ni] = __builtin_amdgcn_wmma_f32_16x16x32_bf16(
                    false, a[mi].v, false, bfr[ni].v,
                    (short)0, acc[mi][ni], false, false);
    }

    // C/D f32 16x16 layout: VGPR r -> row r (lanes 0-15) / row 8+r (16-31),
    // col = lane & 15.
    const int rowSel = half << 3;   // 0 or 8
#pragma unroll
    for (int mi = 0; mi < 2; ++mi) {
#pragma unroll
        for (int ni = 0; ni < 4; ++ni) {
            const int col = nBase + ni * 16 + lid;
            const float bv = bias[col];
            const int row0 = mBase + mi * 16 + rowSel;
#pragma unroll
            for (int r = 0; r < 8; ++r) {
                float v = acc[mi][ni][r] + bv;
                const size_t idx = (size_t)(row0 + r) * N + col;
                if (RELU_BF16_OUT) {
                    v = fmaxf(v, 0.0f);
                    ((bf16_t*)Cout)[idx] = (bf16_t)v;
                } else {
                    ((float*)Cout)[idx] = v;
                }
            }
        }
    }
}

// ---------------------------------------------------------------------------
// Launcher
// ---------------------------------------------------------------------------
extern "C" void kernel_launch(void* const* d_in, const int* in_sizes, int n_in,
                              void* d_out, int out_size, void* d_ws, size_t ws_size,
                              hipStream_t stream)
{
    const float* x  = (const float*)d_in[0];  // [B, Nn, D]
    const float* W1 = (const float*)d_in[1];  // [12D, H]
    const float* b1 = (const float*)d_in[2];  // [H]
    const float* W2 = (const float*)d_in[3];  // [H, O]
    const float* b2 = (const float*)d_in[4];  // [O]

    const int H  = in_sizes[2];               // 4096
    const int O  = in_sizes[4];               // 256
    const int K1 = in_sizes[1] / H;           // 12*D = 3072
    const int D  = K1 / 12;                   // 256
    const int Nn = 128;                       // neighborhood size (reference)
    const int B  = in_sizes[0] / (Nn * D);    // 4096

    // Workspace carve-out (all bf16), 256B aligned.
    size_t off = 0;
    auto carve = [&](size_t bytes) -> void* {
        void* p = (char*)d_ws + off;
        off += (bytes + 255) & ~(size_t)255;
        return p;
    };
    bf16_t* hbf = (bf16_t*)carve((size_t)B * K1 * sizeof(bf16_t)); // [B, 12D]
    bf16_t* W1T = (bf16_t*)carve((size_t)H * K1 * sizeof(bf16_t)); // [H, 12D]
    bf16_t* W2T = (bf16_t*)carve((size_t)O * H  * sizeof(bf16_t)); // [O, H]
    bf16_t* act = (bf16_t*)carve((size_t)B * H  * sizeof(bf16_t)); // [B, H]
    (void)ws_size;

    // 1) aggregation: x -> h (bf16)
    pna_aggregate<<<B, D, 0, stream>>>(x, hbf, Nn, D);

    // 2) weight transpose + convert (goes resident in L2)
    transpose_cvt_bf16<<<dim3(H / 32, K1 / 32), dim3(32, 8), 0, stream>>>(W1, W1T, K1, H);
    transpose_cvt_bf16<<<dim3(O / 32, H  / 32), dim3(32, 8), 0, stream>>>(W2, W2T, H, O);

    // 3) GEMM1 + bias + ReLU -> bf16 activations
    gemm_bf16_wmma<true><<<dim3(B / 64, H / 256), 256, 0, stream>>>(
        hbf, W1T, b1, (void*)act, B, H, K1);

    // 4) GEMM2 + bias -> f32 output
    gemm_bf16_wmma<false><<<dim3(B / 64, O / 256), 256, 0, stream>>>(
        act, W2T, b2, d_out, B, O, H);
}
